// Model_23880018165985
// MI455X (gfx1250) — matrix operand order repaired and checked by reading, not compiled, not run on hardware
//
#include <hip/hip_runtime.h>
#include <hip/hip_bf16.h>

typedef __attribute__((ext_vector_type(16))) _Float16 v16h;
typedef __attribute__((ext_vector_type(8)))  _Float16 v8h;
typedef __attribute__((ext_vector_type(8)))  float    v8f;
typedef __attribute__((ext_vector_type(4)))  unsigned int u32x4;
typedef __attribute__((ext_vector_type(8)))  int      i32x8;
typedef __attribute__((ext_vector_type(4)))  int      i32x4;

#define H_   16
#define HD_  128
#define S_   2048
#define B_   2
#define D_   192
#define NT_  (B_*S_)                      // 4096 tokens
#define QKV_ELEMS ((size_t)B_*H_*S_*HD_)  // 8388608 halfs per tensor

// ---------------- WMMA fragment helpers (wave32, 16x16x32 f16) ----------------
// A-matrix 16x32: lane m = L%16, half = L/16; elem e<8 -> K=8*half+e, e>=8 -> K=16+8*half+(e-8)
__device__ __forceinline__ v16h afrag_f32(const float* row, int k0, int half) {
  const float* p0 = row + k0 + 8 * half;
  const float* p1 = row + k0 + 16 + 8 * half;
  v16h f;
#pragma unroll
  for (int e = 0; e < 8; ++e) { f[e] = (_Float16)p0[e]; f[e + 8] = (_Float16)p1[e]; }
  return f;
}
__device__ __forceinline__ v16h afrag_h(const _Float16* row, int k0, int half) {
  v8h a = *(const v8h*)(row + k0 + 8 * half);
  v8h b = *(const v8h*)(row + k0 + 16 + 8 * half);
  v16h f;
#pragma unroll
  for (int e = 0; e < 8; ++e) { f[e] = a[e]; f[e + 8] = b[e]; }
  return f;
}
// B-matrix 32x16: lane n = L%16, half = L/16; elem e<8 -> K=8*half+e, e>=8 -> K=16+8*half+(e-8)
__device__ __forceinline__ v16h bfrag_f32(const float* row, int k0, int half) {
  const float* p0 = row + k0 + 8 * half;
  const float* p1 = row + k0 + 16 + 8 * half;
  v16h f;
#pragma unroll
  for (int e = 0; e < 8; ++e) { f[e] = (_Float16)p0[e]; f[e + 8] = (_Float16)p1[e]; }
  return f;
}
__device__ __forceinline__ v16h bfrag_h(const _Float16* row, int k0, int half) {
  v8h a = *(const v8h*)(row + k0 + 8 * half);
  v8h b = *(const v8h*)(row + k0 + 16 + 8 * half);
  v16h f;
#pragma unroll
  for (int e = 0; e < 8; ++e) { f[e] = a[e]; f[e + 8] = b[e]; }
  return f;
}
__device__ __forceinline__ v8f wmma_f16(v16h a, v16h b, v8f c) {
  return __builtin_amdgcn_wmma_f32_16x16x32_f16(false, a, false, b, (short)0, c, false, false);
}

// ---------------- Tensor Data Mover: 2D tile (global f16 -> LDS) ----------------
// D# per CDNA5 ISA ch.10.8: group0 = {flags, lds_addr, global_addr[56:0], type=2}
//                           group1 = {mask/data_size, tensor dims, tile dims, dim0 stride}
// Dims/strides in data_size (2-byte) units. Groups 2/3 (and trailing group) zero: 2D tile.
// amdgpu-toolchain (clang-23) 6-arg form: (u32x4, i32x8, i32x4, i32x4, i32x8, i32 cpol).
__device__ __forceinline__ void tdm_load_2d(unsigned int lds_addr, const void* gptr,
                                            unsigned int tensor_d0, unsigned int tensor_d1,
                                            unsigned int tile_d0, unsigned int tile_d1,
                                            unsigned int stride0) {
  unsigned long long ga = (unsigned long long)gptr;
  u32x4 g0;
  g0[0] = 1u;                                             // count=1 (valid), user mode
  g0[1] = lds_addr;                                       // LDS byte address
  g0[2] = (unsigned int)(ga & 0xFFFFFFFFu);               // global_addr[31:0]
  g0[3] = (unsigned int)((ga >> 32) & 0x1FFFFFFu) | (2u << 30);  // addr[56:32], type=2
  i32x8 g1;
  g1[0] = (int)(1u << 16);                                // workgroup_mask=0, data_size=1 (2B)
  g1[1] = (int)((tensor_d0 & 0xFFFFu) << 16);             // tensor_dim0[15:0] @ bit48
  g1[2] = (int)((tensor_d0 >> 16) | ((tensor_d1 & 0xFFFFu) << 16));
  g1[3] = (int)((tensor_d1 >> 16) | (tile_d0 << 16));     // tile_dim0 @ bit112
  g1[4] = (int)tile_d1;                                   // tile_dim1; tile_dim2=0
  g1[5] = (int)stride0;                                   // tensor_dim0_stride[31:0]
  g1[6] = 0;                                              // stride0 hi / stride1 lo
  g1[7] = 0;
  i32x4 gz4 = {};
  i32x8 gz8 = {};
  __builtin_amdgcn_tensor_load_to_lds(g0, g1, gz4, gz4, gz8, 0);
}

// ============ Kernel 1: QKV projection + per-head RMSNorm + RoPE ============
// grid: (NT/64, H, 3[q/k/v]); block: 128 (4 waves); wave: 16 tokens x 128 cols (one head)
__global__ __launch_bounds__(128)
void qkv_kernel(const float* __restrict__ x,
                const float* __restrict__ wq, const float* __restrict__ wk,
                const float* __restrict__ wvp,
                const float* __restrict__ qnw, const float* __restrict__ knw,
                _Float16* __restrict__ qh, _Float16* __restrict__ kh,
                _Float16* __restrict__ vt) {
  const int lane = threadIdx.x & 31, wid = threadIdx.x >> 5;
  const int half = lane >> 4, c = lane & 15;
  const int m0 = blockIdx.x * 64 + wid * 16;
  const int head = blockIdx.y;
  const int z = blockIdx.z;                    // 0=q 1=k 2=v
  const float* w = (z == 0) ? wq : (z == 1) ? wk : wvp;

  v8f zero = {};
  v8f acc[8];
#pragma unroll
  for (int i = 0; i < 8; ++i) acc[i] = zero;

  const float* xrow = x + (size_t)(m0 + c) * D_;
#pragma unroll
  for (int ks = 0; ks < 6; ++ks) {                 // K = 192 = 6 * 32
    v16h a = afrag_f32(xrow, ks * 32, half);
#pragma unroll
    for (int nt = 0; nt < 8; ++nt) {
      const float* wrow = w + (size_t)(head * HD_ + nt * 16 + c) * D_;
      v16h b = bfrag_f32(wrow, ks * 32, half);
      acc[nt] = wmma_f16(a, b, acc[nt]);
    }
  }
  // C layout: acc[nt][r] = out[m = r + 8*half][d = nt*16 + c]
  const int bb = m0 >> 11;
  const int sbase = (m0 & (S_ - 1)) + 8 * half;

  if (z == 2) {
    _Float16* base = vt + (size_t)(bb * H_ + head) * HD_ * S_;
#pragma unroll
    for (int nt = 0; nt < 8; ++nt) {
      int d = nt * 16 + c;
      v8h pk;
#pragma unroll
      for (int r = 0; r < 8; ++r) pk[r] = (_Float16)acc[nt][r];
      *(v8h*)(base + (size_t)d * S_ + sbase) = pk;
    }
    return;
  }
  const float* nw = (z == 0) ? qnw : knw;
  float wn[8];
#pragma unroll
  for (int nt = 0; nt < 8; ++nt) wn[nt] = nw[nt * 16 + c];
  float invf[4];
#pragma unroll
  for (int nt = 0; nt < 4; ++nt) {
    float d = (float)(nt * 16 + c);
    invf[nt] = exp2f(-d * (19.931568569324174f / 64.0f));  // log2(1e6)/64
  }
  _Float16* dst = ((z == 0) ? qh : kh) + (size_t)(bb * H_ + head) * S_ * HD_;
#pragma unroll
  for (int r = 0; r < 8; ++r) {
    float ss = 0.0f;
#pragma unroll
    for (int nt = 0; nt < 8; ++nt) ss += acc[nt][r] * acc[nt][r];
#pragma unroll
    for (int msk = 1; msk < 16; msk <<= 1) ss += __shfl_xor(ss, msk, 32);
    float rms = rsqrtf(ss * (1.0f / 128.0f) + 1e-6f);
    int s = sbase + r;
    float pos = (float)s;
    _Float16* drow = dst + (size_t)s * HD_;
#pragma unroll
    for (int nt = 0; nt < 4; ++nt) {
      float lo = acc[nt][r] * rms * wn[nt];
      float hi = acc[nt + 4][r] * rms * wn[nt + 4];
      float sn, cs;
      __sincosf(pos * invf[nt], &sn, &cs);
      drow[nt * 16 + c]       = (_Float16)(lo * cs - hi * sn);
      drow[(nt + 4) * 16 + c] = (_Float16)(hi * cs + lo * sn);
    }
  }
}

// ============ Kernel 2: flash attention with TDM-staged K/V tiles ============
// grid: (S/64, B*H); block: 128 (4 waves). Wave 0 double-buffers K/V tiles into LDS
// via tensor_load_to_lds; all waves consume fragments from LDS.
__global__ __launch_bounds__(128)
void attn_kernel(const _Float16* __restrict__ qh, const _Float16* __restrict__ kh,
                 const _Float16* __restrict__ vt, const float* __restrict__ mask,
                 _Float16* __restrict__ oh) {
  __shared__ __align__(16) _Float16 kbuf[2][32][128];   // [buf][key][d]   8KB each
  __shared__ __align__(16) _Float16 vbuf[2][128][32];   // [buf][d][key]   8KB each
  __shared__ __align__(16) _Float16 plds[4][16][32];    // per-wave P transpose slab

  const int lane = threadIdx.x & 31, wid = threadIdx.x >> 5;
  const int half = lane >> 4, c = lane & 15;
  const int bh = blockIdx.y;
  const int bb = bh >> 4;
  const int q0 = blockIdx.x * 64 + wid * 16;
  const bool loader = (wid == 0);

  const _Float16* qbase = qh + (size_t)bh * S_ * HD_;
  const _Float16* kbase = kh + (size_t)bh * S_ * HD_;
  const _Float16* vbase = vt + (size_t)bh * HD_ * S_;
  const float*    mbase = mask + (size_t)bb * S_ * S_;

  v16h qf[4];
  const _Float16* qrow = qbase + (size_t)(q0 + c) * HD_;
#pragma unroll
  for (int ks = 0; ks < 4; ++ks) qf[ks] = afrag_h(qrow, ks * 32, half);

  v8f zero = {};
  v8f o[8];
#pragma unroll
  for (int i = 0; i < 8; ++i) o[i] = zero;
  float mrow[8], lrow[8];
#pragma unroll
  for (int r = 0; r < 8; ++r) { mrow[r] = -1e30f; lrow[r] = 0.0f; }

  const float scale = 0.08838834764831845f;   // HD^-0.5
  const int NTILES = S_ / 32;

  // prologue: stage tile 0 into buffer 0
  if (loader) {
    tdm_load_2d((unsigned int)(size_t)&kbuf[0][0][0], kbase,
                /*tensor_d0=*/HD_, /*tensor_d1=*/32, /*tile_d0=*/HD_, /*tile_d1=*/32,
                /*stride0=*/HD_);
    tdm_load_2d((unsigned int)(size_t)&vbuf[0][0][0], vbase,
                /*tensor_d0=*/32, /*tensor_d1=*/HD_, /*tile_d0=*/32, /*tile_d1=*/HD_,
                /*stride0=*/S_);
    __builtin_amdgcn_s_wait_tensorcnt(0);
  }
  __syncthreads();

#pragma unroll 1
  for (int kt = 0; kt < NTILES; ++kt) {
    const int cur = kt & 1, nxt = cur ^ 1;
    const int key0 = kt * 32;
    if (loader && (kt + 1 < NTILES)) {        // prefetch next tile into other buffer
      const int kn = key0 + 32;
      tdm_load_2d((unsigned int)(size_t)&kbuf[nxt][0][0], kbase + (size_t)kn * HD_,
                  HD_, 32, HD_, 32, HD_);
      tdm_load_2d((unsigned int)(size_t)&vbuf[nxt][0][0], vbase + kn,
                  32, HD_, 32, HD_, S_);
    }

    v8f sc[2]; sc[0] = zero; sc[1] = zero;
#pragma unroll
    for (int ks = 0; ks < 4; ++ks) {
#pragma unroll
      for (int nt = 0; nt < 2; ++nt) {
        v16h kf = bfrag_h(&kbuf[cur][nt * 16 + c][0], ks * 32, half);
        sc[nt] = wmma_f16(qf[ks], kf, sc[nt]);
      }
    }
    // scale + additive mask, online softmax per row (rows live in half-wave lane groups)
#pragma unroll
    for (int r = 0; r < 8; ++r) {
      const int mrel = r + 8 * half;
      const float* mr = mbase + (size_t)(q0 + mrel) * S_ + key0;
      float s0 = sc[0][r] * scale + mr[c];
      float s1 = sc[1][r] * scale + mr[16 + c];
      float tmax = fmaxf(s0, s1);
#pragma unroll
      for (int msk = 1; msk < 16; msk <<= 1) tmax = fmaxf(tmax, __shfl_xor(tmax, msk, 32));
      float mnew = fmaxf(mrow[r], tmax);
      float p0 = __expf(s0 - mnew);
      float p1 = __expf(s1 - mnew);
      float psum = p0 + p1;
#pragma unroll
      for (int msk = 1; msk < 16; msk <<= 1) psum += __shfl_xor(psum, msk, 32);
      float corr = __expf(mrow[r] - mnew);
      lrow[r] = lrow[r] * corr + psum;
      mrow[r] = mnew;
#pragma unroll
      for (int dt = 0; dt < 8; ++dt) o[dt][r] *= corr;
      plds[wid][mrel][c]      = (_Float16)p0;
      plds[wid][mrel][16 + c] = (_Float16)p1;
    }
    v16h pf = afrag_h(&plds[wid][c][0], 0, half);
#pragma unroll
    for (int dt = 0; dt < 8; ++dt) {
      v16h vf = bfrag_h(&vbuf[cur][dt * 16 + c][0], 0, half);
      o[dt] = wmma_f16(pf, vf, o[dt]);
    }

    if (loader && (kt + 1 < NTILES)) __builtin_amdgcn_s_wait_tensorcnt(0);
    __syncthreads();   // publish next buffer; also guards WAR on the buffer we just read
  }

  _Float16* obase = oh + (size_t)bh * S_ * HD_;
#pragma unroll
  for (int r = 0; r < 8; ++r) {
    float inv = 1.0f / lrow[r];
    _Float16* orow = obase + (size_t)(q0 + r + 8 * half) * HD_;
#pragma unroll
    for (int dt = 0; dt < 8; ++dt) orow[dt * 16 + c] = (_Float16)(o[dt][r] * inv);
  }
}

// ============ Kernel 3: output projection out = attn @ wo^T ============
__global__ __launch_bounds__(128)
void proj_kernel(const _Float16* __restrict__ oh, const float* __restrict__ wo,
                 float* __restrict__ out) {
  const int lane = threadIdx.x & 31, wid = threadIdx.x >> 5;
  const int half = lane >> 4, c = lane & 15;
  const int m0 = blockIdx.x * 64 + wid * 16;
  const int bb = m0 >> 11;
  const int srow = (m0 & (S_ - 1)) + c;

  v8f zero = {};
  v8f acc[12];
#pragma unroll
  for (int i = 0; i < 12; ++i) acc[i] = zero;

#pragma unroll 1
  for (int ks = 0; ks < 64; ++ks) {
    const int h = ks >> 2;
    const int d0 = (ks & 3) * 32;
    const _Float16* arow = oh + ((size_t)(bb * H_ + h) * S_ + srow) * HD_ + d0;
    v16h a = afrag_h(arow, 0, half);
#pragma unroll
    for (int nt = 0; nt < 12; ++nt) {
      const float* wrow = wo + (size_t)(nt * 16 + c) * (H_ * HD_);
      v16h b = bfrag_f32(wrow, ks * 32, half);
      acc[nt] = wmma_f16(a, b, acc[nt]);
    }
  }
#pragma unroll
  for (int r = 0; r < 8; ++r) {
    float* orow = out + (size_t)(m0 + r + 8 * half) * D_;
#pragma unroll
    for (int nt = 0; nt < 12; ++nt) orow[nt * 16 + c] = acc[nt][r];
  }
}

extern "C" void kernel_launch(void* const* d_in, const int* in_sizes, int n_in,
                              void* d_out, int out_size, void* d_ws, size_t ws_size,
                              hipStream_t stream) {
  (void)in_sizes; (void)n_in; (void)out_size; (void)ws_size;
  const float* x    = (const float*)d_in[0];
  const float* mask = (const float*)d_in[1];
  const float* wq   = (const float*)d_in[2];
  const float* wk   = (const float*)d_in[3];
  const float* wv   = (const float*)d_in[4];
  const float* wo   = (const float*)d_in[5];
  const float* qnw  = (const float*)d_in[6];
  const float* knw  = (const float*)d_in[7];
  float* out = (float*)d_out;

  _Float16* ws = (_Float16*)d_ws;           // 4 x 16 MB f16 tensors = 64 MB scratch
  _Float16* qh = ws;
  _Float16* kh = ws + QKV_ELEMS;
  _Float16* vt = ws + 2 * QKV_ELEMS;
  _Float16* oh = ws + 3 * QKV_ELEMS;

  qkv_kernel<<<dim3(NT_ / 64, H_, 3), 128, 0, stream>>>(x, wq, wk, wv, qnw, knw, qh, kh, vt);
  attn_kernel<<<dim3(S_ / 64, B_ * H_), 128, 0, stream>>>(qh, kh, vt, mask, oh);
  proj_kernel<<<dim3(NT_ / 64), 128, 0, stream>>>(oh, wo, out);
}
